// MaskedOmics_5325759447053
// MI455X (gfx1250) — matrix operand order repaired and checked by reading, not compiled, not run on hardware
//
#include <hip/hip_runtime.h>
#include <hip/hip_bf16.h>

// ---- problem constants (match reference) ----
#define B_   4096
#define IN_  4096
#define NP_  512
#define DP1_ 8
#define DP2_ 16
#define P1_  4096   // NP*DP1
#define P2_  8192   // NP*DP2
#define H_   2048   // P2/4
#define NC_  4

typedef __attribute__((ext_vector_type(16))) __bf16 v16bf;
typedef __attribute__((ext_vector_type(8)))  float  v8f;

union Frag { uint4 u[2]; v16bf v; };

// ---- CDNA5 async global->LDS copy (ASYNCcnt-tracked DMA path) ----
__device__ __forceinline__ void async_load_b128(const void* gptr, void* lds_ptr) {
    // Low 32 bits of the flat address of a __shared__ object are the LDS offset.
    unsigned lds_off = (unsigned)(uintptr_t)lds_ptr;
    asm volatile("global_load_async_to_lds_b128 %0, %1, off"
                 :: "v"(lds_off), "v"(gptr)
                 : "memory");
}
__device__ __forceinline__ void wait_async_zero() {
    asm volatile("s_wait_asynccnt 0" ::: "memory");
}

// =====================================================================
// Prep: tiled transpose f32[R][C] -> bf16 dst[C][R], optional fc1 mask
//   (mask1[i,j] = comp[i, j/8], i = source row, j = source col)
// =====================================================================
template <bool MASK>
__global__ __launch_bounds__(256)
void transpose_bf16_kernel(const float* __restrict__ src, const int* __restrict__ comp,
                           __bf16* __restrict__ dst, int R, int C) {
    __shared__ __bf16 t[64][72];          // padded: row stride 144B
    const int tid = threadIdx.x;
    const int r0 = blockIdx.y * 64;
    const int c0 = blockIdx.x * 64;
    {
        int r  = tid >> 2;                // 0..63
        int cc = (tid & 3) * 16;          // 0,16,32,48
        const float* sp = src + (size_t)(r0 + r) * C + (c0 + cc);
        #pragma unroll
        for (int x = 0; x < 16; x += 4) {
            float4 v = *(const float4*)(sp + x);
            float m0 = 1.f, m1 = 1.f, m2 = 1.f, m3 = 1.f;
            if (MASK) {
                int j = c0 + cc + x;
                const int* cr = comp + (size_t)(r0 + r) * NP_;
                m0 = cr[(j + 0) >> 3] ? 1.f : 0.f;
                m1 = cr[(j + 1) >> 3] ? 1.f : 0.f;
                m2 = cr[(j + 2) >> 3] ? 1.f : 0.f;
                m3 = cr[(j + 3) >> 3] ? 1.f : 0.f;
            }
            t[cc + x + 0][r] = (__bf16)(v.x * m0);
            t[cc + x + 1][r] = (__bf16)(v.y * m1);
            t[cc + x + 2][r] = (__bf16)(v.z * m2);
            t[cc + x + 3][r] = (__bf16)(v.w * m3);
        }
    }
    __syncthreads();
    {
        int c  = tid >> 2;                // 0..63
        int rr = (tid & 3) * 16;          // 0,16,32,48
        __bf16* dp = dst + (size_t)(c0 + c) * R + (r0 + rr);
        *(uint4*)(dp + 0) = *(const uint4*)&t[c][rr + 0];
        *(uint4*)(dp + 8) = *(const uint4*)&t[c][rr + 8];
    }
}

// =====================================================================
// Tiled WMMA GEMM:  C[M,N] = A[M,K] @ B[K,N] + bias (optional ReLU)
//   A: f32 or bf16 row-major; B supplied TRANSPOSED: BT[N][K] bf16.
//   Block tile 64(M) x 128(N), K-step 32, double-buffered LDS,
//   8 waves (2x4), each wave computes 32x32 = 4 WMMA accumulators.
//   B-tile (and bf16 A-tile) staging uses async global->LDS DMA.
// =====================================================================
template <bool A_IS_F32, bool RELU>
__global__ __launch_bounds__(256)
void gemm_wmma_bf16(const void* __restrict__ Aptr, const __bf16* __restrict__ BT,
                    const float* __restrict__ bias, __bf16* __restrict__ C,
                    int M, int N, int K) {
    __shared__ __bf16 As[2][64][40];      // [m][k], padded rows (80B)
    __shared__ __bf16 Bs[2][128][40];     // [n][k], padded rows (80B)

    const int tid   = threadIdx.x;
    const int lane  = tid & 31;
    const int wave  = tid >> 5;
    const int wm    = wave >> 2;          // 0..1 -> M offset wm*32
    const int wn    = wave & 3;           // 0..3 -> N offset wn*32
    const int bm    = blockIdx.y * 64;
    const int bn    = blockIdx.x * 128;
    const int row16 = lane & 15;
    const int hi    = (lane >> 4) & 1;

    v8f acc[2][2] = {};

    auto stage = [&](int buf, int k0) {
        // A tile 64x32: thread -> row tid/4, 8 consecutive k
        {
            int r = tid >> 2;             // 0..63
            int c = (tid & 3) * 8;        // 0,8,16,24
            if constexpr (A_IS_F32) {
                const float* Af = (const float*)Aptr;
                const float4* p = (const float4*)(Af + (size_t)(bm + r) * K + (k0 + c));
                float4 v0 = p[0], v1 = p[1];
                union { uint4 u; __bf16 h[8]; } pk;
                pk.h[0] = (__bf16)v0.x; pk.h[1] = (__bf16)v0.y;
                pk.h[2] = (__bf16)v0.z; pk.h[3] = (__bf16)v0.w;
                pk.h[4] = (__bf16)v1.x; pk.h[5] = (__bf16)v1.y;
                pk.h[6] = (__bf16)v1.z; pk.h[7] = (__bf16)v1.w;
                *(uint4*)&As[buf][r][c] = pk.u;
            } else {
                const __bf16* Ab = (const __bf16*)Aptr;
                async_load_b128(Ab + (size_t)(bm + r) * K + (k0 + c), &As[buf][r][c]);
            }
        }
        // B tile (natural copy of BT): 128 rows x 32 k, thread -> row tid/2
        {
            int n = tid >> 1;             // 0..127
            int c = (tid & 1) * 16;       // 0 or 16
            const __bf16* g = BT + (size_t)(bn + n) * K + (k0 + c);
            async_load_b128(g + 0, &Bs[buf][n][c + 0]);
            async_load_b128(g + 8, &Bs[buf][n][c + 8]);
        }
    };

    auto compute = [&](int buf) {
        Frag a[2], b[2];
        #pragma unroll
        for (int t = 0; t < 2; ++t) {
            // A frag (16x32): per-lane K runs {hi*8+0..7} and {16+hi*8+0..7}
            const __bf16* ar = As[buf][wm * 32 + t * 16 + row16];
            a[t].u[0] = *(const uint4*)(ar + hi * 8);
            a[t].u[1] = *(const uint4*)(ar + 16 + hi * 8);
            // B frag (32x16): per-lane K runs {hi*8+0..7} and {16+hi*8+0..7}
            const __bf16* br = Bs[buf][wn * 32 + t * 16 + row16];
            b[t].u[0] = *(const uint4*)(br + hi * 8);
            b[t].u[1] = *(const uint4*)(br + 16 + hi * 8);
        }
        #pragma unroll
        for (int i = 0; i < 2; ++i)
            #pragma unroll
            for (int j = 0; j < 2; ++j)
                acc[i][j] = __builtin_amdgcn_wmma_f32_16x16x32_bf16(
                    false, a[i].v, false, b[j].v, (short)0, acc[i][j], false, false);
    };

    const int nk = K >> 5;
    stage(0, 0);
    wait_async_zero();
    __syncthreads();
    for (int kt = 0; kt < nk; ++kt) {
        if (kt + 1 < nk) stage((kt + 1) & 1, (kt + 1) << 5);  // async DMA overlaps compute
        compute(kt & 1);
        wait_async_zero();
        __syncthreads();
    }

    // epilogue: C/D layout -> lanes 0-15: N=lane, M=r ; lanes 16-31: M=8+r
    #pragma unroll
    for (int i = 0; i < 2; ++i) {
        #pragma unroll
        for (int j = 0; j < 2; ++j) {
            int n = bn + wn * 32 + j * 16 + row16;
            float bv = bias[n];
            #pragma unroll
            for (int r = 0; r < 8; ++r) {
                int m = bm + wm * 32 + i * 16 + hi * 8 + r;
                float v = acc[i][j][r] + bv;
                if (RELU) v = fmaxf(v, 0.0f);
                C[(size_t)m * N + n] = (__bf16)v;
            }
        }
    }
}

// =====================================================================
// Block-diagonal pathway layer:
//   out2[b, p*16+j] = sum_i out1[b,p*8+i] * fc2_w[p*8+i, p*16+j] + fc2_b
// =====================================================================
__global__ __launch_bounds__(256)
void gemm2_blockdiag_kernel(const __bf16* __restrict__ out1, const float* __restrict__ fc2_w,
                            const float* __restrict__ fc2_b, __bf16* __restrict__ out2) {
    size_t idx = (size_t)blockIdx.x * 256 + threadIdx.x;   // over B_*P2_
    int col = (int)(idx & (P2_ - 1));
    int b   = (int)(idx >> 13);                            // P2_ = 2^13
    int p   = col >> 4;                                    // pathway (DP2_=16)
    float s = fc2_b[col];
    const __bf16* a = out1 + (size_t)b * P1_ + p * DP1_;
    const float*  w = fc2_w + (size_t)(p * DP1_) * P2_ + col;
    #pragma unroll
    for (int i = 0; i < DP1_; ++i)
        s += (float)a[i] * w[(size_t)i * P2_];
    out2[idx] = (__bf16)s;
}

// =====================================================================
// logits[b,c] = h[b,:] . w2[:,c] + b2[c]    (NC_=4)
// =====================================================================
__global__ __launch_bounds__(256)
void logits_kernel(const __bf16* __restrict__ h, const float* __restrict__ w2,
                   const float* __restrict__ b2, float* __restrict__ logits) {
    int idx = blockIdx.x * 256 + threadIdx.x;              // b*4+c
    int c = idx & 3;
    int b = idx >> 2;
    float s = b2[c];
    const __bf16* hr = h + (size_t)b * H_;
    #pragma unroll 4
    for (int k = 0; k < H_; ++k)
        s += (float)hr[k] * w2[(size_t)k * NC_ + c];
    logits[idx] = s;
}

// =====================================================================
// Finalize (reference's axis=1 on [1,B,NC]):
//   hazards = sigmoid; S = cumprod over batch; Y_hat = argmax over batch
// =====================================================================
__global__ void finalize_kernel(const float* __restrict__ logits, float* __restrict__ out) {
    int c = threadIdx.x;
    if (c >= NC_) return;
    float prod = 1.0f;
    float best = -__builtin_huge_valf();
    int bestb = 0;
    for (int b = 0; b < B_; ++b) {
        float lg = logits[b * NC_ + c];
        if (lg > best) { best = lg; bestb = b; }
        float hz = 1.0f / (1.0f + expf(-lg));
        out[b * NC_ + c] = hz;                       // hazards [1,B,NC]
        prod *= (1.0f - hz);
        out[B_ * NC_ + b * NC_ + c] = prod;          // S [1,B,NC]
    }
    out[2 * B_ * NC_ + c] = (float)bestb;            // Y_hat [1,1,NC]
}

// =====================================================================
extern "C" void kernel_launch(void* const* d_in, const int* in_sizes, int n_in,
                              void* d_out, int out_size, void* d_ws, size_t ws_size,
                              hipStream_t stream) {
    const float* x_omic = (const float*)d_in[0];   // [B,IN]
    const float* fc1_w  = (const float*)d_in[1];   // [IN,P1]
    const float* fc1_b  = (const float*)d_in[2];   // [P1]
    const float* fc2_w  = (const float*)d_in[3];   // [P1,P2]
    const float* fc2_b  = (const float*)d_in[4];   // [P2]
    const float* w1     = (const float*)d_in[5];   // [P2,H]
    const float* b1     = (const float*)d_in[6];   // [H]
    const float* w2     = (const float*)d_in[7];   // [H,NC]
    const float* b2     = (const float*)d_in[8];   // [NC]
    const int*   comp   = (const int*)d_in[9];     // [IN,NP]
    float* out = (float*)d_out;

    // workspace layout (bf16 elements)
    __bf16* ws    = (__bf16*)d_ws;
    __bf16* w1mT  = ws;                              // [P1][IN]  16M (transposed, masked)
    __bf16* out1  = ws + (size_t)16 * 1024 * 1024;   // [B][P1]   16M
    __bf16* out2  = ws + (size_t)32 * 1024 * 1024;   // [B][P2]   32M
    __bf16* w1T   = ws + (size_t)64 * 1024 * 1024;   // [H][P2]   16M (transposed)
    __bf16* hbuf  = ws;                              // [B][H]    8M  (reuses w1mT region)
    float*  logits = (float*)(ws + (size_t)80 * 1024 * 1024);  // [B,NC]

    // 1) weight prep: mask+convert+transpose fc1_w; convert+transpose w1
    transpose_bf16_kernel<true><<<dim3(P1_ / 64, IN_ / 64), 256, 0, stream>>>(
        fc1_w, comp, w1mT, IN_, P1_);
    transpose_bf16_kernel<false><<<dim3(H_ / 64, P2_ / 64), 256, 0, stream>>>(
        w1, nullptr, w1T, P2_, H_);

    // 2) GEMM1: out1 = x_omic @ (fc1_w*mask1) + fc1_b    (M=B, N=P1, K=IN)
    gemm_wmma_bf16<true, false><<<dim3(P1_ / 128, B_ / 64), 256, 0, stream>>>(
        (const void*)x_omic, w1mT, fc1_b, out1, B_, P1_, IN_);

    // 3) block-diagonal layer: out2 = out1 @ (fc2_w*mask2) + fc2_b
    gemm2_blockdiag_kernel<<<(size_t)B_ * P2_ / 256, 256, 0, stream>>>(out1, fc2_w, fc2_b, out2);

    // 4) GEMM3: h = relu(out2 @ w1 + b1)                 (M=B, N=H, K=P2)
    gemm_wmma_bf16<false, true><<<dim3(H_ / 128, B_ / 64), 256, 0, stream>>>(
        (const void*)out2, w1T, b1, hbuf, B_, H_, P2_);

    // 5) logits = h @ w2 + b2
    logits_kernel<<<(B_ * NC_) / 256, 256, 0, stream>>>(hbuf, w2, b2, logits);

    // 6) sigmoid + batch cumprod + batch argmax
    finalize_kernel<<<1, NC_, 0, stream>>>(logits, out);
}
